// CosineAttention_54082228191953
// MI455X (gfx1250) — matrix operand order repaired and checked by reading, not compiled, not run on hardware
//
#include <hip/hip_runtime.h>

// ---------------------------------------------------------------------------
// Cosine attention for MI455X (gfx1250, wave32, WMMA f32_16x16x32_f16 + TDM)
//   B=2, N=2048, C=512, H=8, D=64
// ---------------------------------------------------------------------------

typedef __attribute__((ext_vector_type(16))) _Float16 v16h;
typedef __attribute__((ext_vector_type(8)))  float    v8f;
typedef __attribute__((ext_vector_type(4)))  unsigned u32x4;
typedef __attribute__((ext_vector_type(8)))  int      i32x8;
typedef __attribute__((ext_vector_type(4)))  int      i32x4;

#define Bb   2
#define Nn   2048
#define Cc   512
#define Hh   8
#define Dd   64
#define QKVC 1536          // 3*H*D
#define HD   512           // H*D
#define NEG_MAX 3.402823466e38f

#define WMMA_F16(a, b, c) \
  __builtin_amdgcn_wmma_f32_16x16x32_f16(false, (a), false, (b), (short)0, (c), false, false)

// Tensor Data Mover availability (device pass only; host pass falls back)
#if defined(__HIP_DEVICE_COMPILE__) && __has_builtin(__builtin_amdgcn_tensor_load_to_lds)
#define HAVE_TDM 1
#else
#define HAVE_TDM 0
#endif

// A-fragment loader (16-bit A 16x32, MxK): lane L -> row m = L&15,
// VGPR v holds K = 2v + 16*(L>=16), pairs (k, k+1).
__device__ __forceinline__ v16h load_a_frag(const _Float16* row_k0, int hi) {
  union { v16h h; unsigned u[8]; } A;
  const unsigned* p32 = (const unsigned*)row_k0;
#pragma unroll
  for (int v = 0; v < 8; ++v) {
    int k = 2 * v + hi * 16;
    A.u[v] = p32[k >> 1];
  }
  return A.h;
}

__device__ __forceinline__ v16h load_b_frag(const _Float16* p) {
  return *(const v16h*)p;
}

#if HAVE_TDM
// Issue a TDM 2-D tile load: f32 tile (tile_dim0=32 cols, tile_dim1=16 rows)
// from a row-major tensor with row stride Nn, into LDS at lds_off.
// D# bitfields per CDNA5 ISA.
__device__ __forceinline__ void tdm_load_bias_tile(const float* gaddr, unsigned lds_off,
                                                   int remX, int remY) {
  unsigned long long ga = (unsigned long long)(size_t)gaddr;
  u32x4 g0;
  g0[0] = 1u;                                          // count=1 valid user descriptor
  g0[1] = lds_off;                                     // lds_addr
  g0[2] = (unsigned)(ga & 0xFFFFFFFFu);                // global_addr[31:0]
  g0[3] = (unsigned)((ga >> 32) & 0x1FFFFFFu) | (2u << 30);  // global_addr[56:32] | type=2
  i32x8 g1;
  g1[0] = (2 << 16);                                   // data_size=2 (4 bytes), no multicast
  g1[1] = (remX & 0xFFFF) << 16;                       // tensor_dim0[15:0]
  g1[2] = ((remX >> 16) & 0xFFFF) | ((remY & 0xFFFF) << 16);   // dim0[31:16] | dim1[15:0]
  g1[3] = ((remY >> 16) & 0xFFFF) | (32 << 16);        // dim1[31:16] | tile_dim0=32
  g1[4] = 16;                                          // tile_dim1=16, tile_dim2=0
  g1[5] = Nn;                                          // tensor_dim0_stride[31:0]
  g1[6] = 0;                                           // stride hi | dim1_stride lo
  g1[7] = 0;
  i32x4 g2 = (i32x4)0, g3 = (i32x4)0;
#if __clang_major__ >= 23
  __builtin_amdgcn_tensor_load_to_lds(g0, g1, g2, g3, (i32x8)0, 0);
#else
  __builtin_amdgcn_tensor_load_to_lds(g0, g1, g2, g3, 0);
#endif
}
#endif

// ---------------------------------------------------------------------------
// elementwise f32 -> f16 convert (grid-stride)
__global__ void cvt_f32_f16(const float* __restrict__ s, _Float16* __restrict__ d, long n) {
  long i = (long)blockIdx.x * blockDim.x + threadIdx.x;
  long stride = (long)gridDim.x * blockDim.x;
  for (; i < n; i += stride) d[i] = (_Float16)s[i];
}

// transpose f32[rows][cols] -> f16[cols][rows]
__global__ void transpose_f32_f16(const float* __restrict__ s, _Float16* __restrict__ d,
                                  int rows, int cols) {
  long i = (long)blockIdx.x * blockDim.x + threadIdx.x;
  long total = (long)rows * cols;
  if (i < total) {
    long c = i / rows, r = i % rows;
    d[i] = (_Float16)s[r * (long)cols + c];
  }
}

// ---------------------------------------------------------------------------
// QKV projection: [B*N, C] @ [C, 3HD], de-interleaved store.
// Software-pipelined K loop: next iteration's fragments load while WMMAs run.
__global__ __launch_bounds__(128)
void qkv_gemm(const _Float16* __restrict__ xh, const _Float16* __restrict__ wT,
              float* __restrict__ Qf, float* __restrict__ Kf,
              _Float16* __restrict__ Vt) {
  const int lane = threadIdx.x & 31, wave = threadIdx.x >> 5;
  const int hi = lane >> 4, ln = lane & 15;
  const int tile = blockIdx.x * 4 + wave;          // 6144 tiles = 256 strips * 24 cgroups
  const int strip = tile & 255;
  const int cg = tile >> 8;
  const int row0 = strip * 16, col0 = cg * 64;

  v8f acc[4];
  const v8f vzero = {0.f, 0.f, 0.f, 0.f, 0.f, 0.f, 0.f, 0.f};
#pragma unroll
  for (int g = 0; g < 4; ++g) acc[g] = vzero;

  const _Float16* arow  = xh + (long)(row0 + ln) * Cc;
  const _Float16* bcol0 = wT + (long)(col0 + ln) * Cc + hi * 16;

  v16h a  = load_a_frag(arow, hi);
  v16h b0 = load_b_frag(bcol0);
  v16h b1 = load_b_frag(bcol0 + 16 * Cc);
  v16h b2 = load_b_frag(bcol0 + 32 * Cc);
  v16h b3 = load_b_frag(bcol0 + 48 * Cc);

  for (int kk = 0; kk < Cc; kk += 32) {
    v16h an = a, bn0 = b0, bn1 = b1, bn2 = b2, bn3 = b3;
    const int kn = kk + 32;
    if (kn < Cc) {                                 // prefetch next K-slice
      an  = load_a_frag(arow + kn, hi);
      bn0 = load_b_frag(bcol0 + kn);
      bn1 = load_b_frag(bcol0 + 16 * Cc + kn);
      bn2 = load_b_frag(bcol0 + 32 * Cc + kn);
      bn3 = load_b_frag(bcol0 + 48 * Cc + kn);
    }
    acc[0] = WMMA_F16(a, b0, acc[0]);
    acc[1] = WMMA_F16(a, b1, acc[1]);
    acc[2] = WMMA_F16(a, b2, acc[2]);
    acc[3] = WMMA_F16(a, b3, acc[3]);
    a = an; b0 = bn0; b1 = bn1; b2 = bn2; b3 = bn3;
  }
#pragma unroll
  for (int g = 0; g < 4; ++g) {
#pragma unroll
    for (int r = 0; r < 8; ++r) {
      int row = row0 + r + hi * 8;                 // [0, B*N)
      int c = col0 + g * 16 + ln;                  // [0, 3HD); layout (h, d, qkv)
      int hd = c / 192, rem = c % 192, d = rem / 3, s = rem % 3;
      int bb = row >> 11, n = row & (Nn - 1);
      float val = acc[g][r];
      long bhn = ((long)(bb * Hh + hd) * Nn + n) * Dd + d;
      if (s == 0)      Qf[bhn] = val;
      else if (s == 1) Kf[bhn] = val;
      else             Vt[((long)(bb * Hh + hd) * Dd + d) * Nn + n] = (_Float16)val;
    }
  }
}

// ---------------------------------------------------------------------------
// per-(b,h,n) L2 normalize; fold temperature into Q.  one wave per row.
__global__ __launch_bounds__(256)
void norm_qk(const float* __restrict__ Qf, const float* __restrict__ Kf,
             const float* __restrict__ temp,
             _Float16* __restrict__ Qh, _Float16* __restrict__ Kh) {
  const int lane = threadIdx.x & 31, wave = threadIdx.x >> 5;
  const long row = (long)blockIdx.x * 8 + wave;    // [0, B*H*N)
  const float* q = Qf + row * Dd;
  const float* k = Kf + row * Dd;
  float q0 = q[lane], q1 = q[lane + 32];
  float k0 = k[lane], k1 = k[lane + 32];
  float sq = q0 * q0 + q1 * q1;
  float sk = k0 * k0 + k1 * k1;
#pragma unroll
  for (int m = 16; m > 0; m >>= 1) {
    sq += __shfl_xor(sq, m, 32);
    sk += __shfl_xor(sk, m, 32);
  }
  float t = *temp;
  float iq = t / fmaxf(sqrtf(sq), 1e-12f);
  float ik = 1.0f / fmaxf(sqrtf(sk), 1e-12f);
  _Float16* qo = Qh + row * Dd;
  _Float16* ko = Kh + row * Dd;
  qo[lane]      = (_Float16)(q0 * iq);
  qo[lane + 32] = (_Float16)(q1 * iq);
  ko[lane]      = (_Float16)(k0 * ik);
  ko[lane + 32] = (_Float16)(k1 * ik);
}

// ---------------------------------------------------------------------------
// Flash-style attention: one wave owns 16 query rows of one (b,h);
// 32 key columns per step.  Vt is [B,H,D,N] f16.
// pos_bias tiles are streamed into LDS by the Tensor Data Mover, double
// buffered one iteration ahead so the 134MB bias stream overlaps WMMA work.
__global__ __launch_bounds__(128)
void attn_kernel(const _Float16* __restrict__ Qh, const _Float16* __restrict__ Kh,
                 const _Float16* __restrict__ Vt, const float* __restrict__ bias,
                 const unsigned char* __restrict__ mask, _Float16* __restrict__ Ah) {
  __shared__ _Float16 P[4][16 * 32];               // per-wave P staging
#if HAVE_TDM
  __shared__ float BiasT[4][2][16 * 32];           // per-wave double-buffered bias tiles
#endif
  const int lane = threadIdx.x & 31, wave = threadIdx.x >> 5;
  const int hi = lane >> 4, ln = lane & 15;
  const int t = blockIdx.x * 4 + wave;             // 2048 strips
  const int q0 = (t & 127) * 16;                   // N/16 strips per (b,h)
  const int bh = t >> 7;
  const int bb = bh / Hh, head = bh % Hh;

  const _Float16* Qp = Qh + ((long)bh * Nn + q0) * Dd;
  const v16h A0 = load_a_frag(Qp + (long)ln * Dd, hi);
  const v16h A1 = load_a_frag(Qp + (long)ln * Dd + 32, hi);

  float mrow[8], lrow[8];
  v8f O[4];
  const v8f vzero = {0.f, 0.f, 0.f, 0.f, 0.f, 0.f, 0.f, 0.f};
#pragma unroll
  for (int g = 0; g < 4; ++g) O[g] = vzero;
#pragma unroll
  for (int r = 0; r < 8; ++r) { mrow[r] = -NEG_MAX; lrow[r] = 0.f; }

  unsigned char qm[8];
#pragma unroll
  for (int r = 0; r < 8; ++r) qm[r] = mask[bb * Nn + q0 + r + hi * 8];

  const _Float16* Kbase = Kh + (long)bh * Nn * Dd;
  const _Float16* Vbase = Vt + (long)bh * Dd * Nn;
  const float* biasrow = bias + (long)head * Nn * Nn + (long)q0 * Nn;

#if HAVE_TDM
  tdm_load_bias_tile(biasrow, (unsigned)(size_t)&BiasT[wave][0][0], Nn, Nn - q0);
#endif

  for (int j = 0; j < Nn; j += 32) {
#if HAVE_TDM
    const int buf = (j >> 5) & 1;
    if (j + 32 < Nn)                               // DMA next bias tile ahead of use
      tdm_load_bias_tile(biasrow + j + 32, (unsigned)(size_t)&BiasT[wave][buf ^ 1][0],
                         Nn - (j + 32), Nn - q0);
#endif
    // --- issue all K and V fragment loads up front (overlap with WMMA/softmax)
    v16h B00 = load_b_frag(Kbase + (long)(j + ln) * Dd + hi * 16);
    v16h B01 = load_b_frag(Kbase + (long)(j + ln) * Dd + 32 + hi * 16);
    v16h B10 = load_b_frag(Kbase + (long)(j + 16 + ln) * Dd + hi * 16);
    v16h B11 = load_b_frag(Kbase + (long)(j + 16 + ln) * Dd + 32 + hi * 16);
    v16h VB[4];
#pragma unroll
    for (int g = 0; g < 4; ++g)
      VB[g] = load_b_frag(Vbase + (long)(g * 16 + ln) * Nn + j + hi * 16);
    unsigned char km0 = mask[bb * Nn + j + ln];
    unsigned char km1 = mask[bb * Nn + j + 16 + ln];

    // --- S = Qn @ Kn^T : two 16x16 tiles, D=64 contraction in two K=32 halves
    v8f S0 = vzero, S1 = vzero;
    S0 = WMMA_F16(A0, B00, S0);
    S0 = WMMA_F16(A1, B01, S0);
    S1 = WMMA_F16(A0, B10, S1);
    S1 = WMMA_F16(A1, B11, S1);

#if HAVE_TDM
    if (j + 32 < Nn) __builtin_amdgcn_s_wait_tensorcnt(1);  // current tile landed
    else             __builtin_amdgcn_s_wait_tensorcnt(0);
#endif

    // --- bias + mask + online softmax (row = r + 8*hi matches C-frag layout)
#pragma unroll
    for (int r = 0; r < 8; ++r) {
#if HAVE_TDM
      float b0 = BiasT[wave][buf][(r + hi * 8) * 32 + ln];
      float b1 = BiasT[wave][buf][(r + hi * 8) * 32 + 16 + ln];
#else
      int row = q0 + r + hi * 8;
      float b0 = biasrow[(long)(r + hi * 8) * Nn + j + ln];
      float b1 = biasrow[(long)(r + hi * 8) * Nn + j + 16 + ln];
      (void)row;
#endif
      float s0 = (qm[r] | km0) ? -NEG_MAX : S0[r] + b0;
      float s1 = (qm[r] | km1) ? -NEG_MAX : S1[r] + b1;
      float tmax = fmaxf(s0, s1);
#pragma unroll
      for (int xm = 8; xm > 0; xm >>= 1) tmax = fmaxf(tmax, __shfl_xor(tmax, xm, 32));
      float mn = fmaxf(mrow[r], tmax);
      float scale = __expf(mrow[r] - mn);
      float p0 = __expf(s0 - mn);
      float p1 = __expf(s1 - mn);
      float rs = p0 + p1;
#pragma unroll
      for (int xm = 8; xm > 0; xm >>= 1) rs += __shfl_xor(rs, xm, 32);
      lrow[r] = lrow[r] * scale + rs;
      mrow[r] = mn;
#pragma unroll
      for (int g = 0; g < 4; ++g) O[g][r] *= scale;
      P[wave][(r + hi * 8) * 32 + ln] = (_Float16)p0;
      P[wave][(r + hi * 8) * 32 + 16 + ln] = (_Float16)p1;
    }
    asm volatile("s_wait_dscnt 0" ::: "memory");   // per-wave LDS RAW fence

    // --- O += P @ V : P reloaded in A-layout, V columns contiguous (transposed)
    v16h PA = load_a_frag(&P[wave][ln * 32], hi);
#pragma unroll
    for (int g = 0; g < 4; ++g) O[g] = WMMA_F16(PA, VB[g], O[g]);
  }

  // epilogue: O / l, write [B*N, H*D] f16
#pragma unroll
  for (int g = 0; g < 4; ++g) {
#pragma unroll
    for (int r = 0; r < 8; ++r) {
      int row = q0 + r + hi * 8;
      Ah[((long)(bb * Nn + row)) * HD + head * Dd + g * 16 + ln] =
          (_Float16)(O[g][r] / lrow[r]);
    }
  }
}

// ---------------------------------------------------------------------------
// output projection: [B*N, HD] @ [HD, C] -> f32 d_out.  WoT: [C][HD] f16.
__global__ __launch_bounds__(128)
void out_gemm(const _Float16* __restrict__ Ah, const _Float16* __restrict__ WoT,
              float* __restrict__ out) {
  const int lane = threadIdx.x & 31, wave = threadIdx.x >> 5;
  const int hi = lane >> 4, ln = lane & 15;
  const int tile = blockIdx.x * 4 + wave;          // 2048 tiles = 256 strips * 8 cgroups
  const int strip = tile & 255;
  const int cg = tile >> 8;
  const int row0 = strip * 16, col0 = cg * 64;

  v8f acc[4];
  const v8f vzero = {0.f, 0.f, 0.f, 0.f, 0.f, 0.f, 0.f, 0.f};
#pragma unroll
  for (int g = 0; g < 4; ++g) acc[g] = vzero;

  const _Float16* arow  = Ah + (long)(row0 + ln) * HD;
  const _Float16* bcol0 = WoT + (long)(col0 + ln) * HD + hi * 16;

  v16h a  = load_a_frag(arow, hi);
  v16h b0 = load_b_frag(bcol0);
  v16h b1 = load_b_frag(bcol0 + 16 * HD);
  v16h b2 = load_b_frag(bcol0 + 32 * HD);
  v16h b3 = load_b_frag(bcol0 + 48 * HD);

  for (int kk = 0; kk < HD; kk += 32) {
    v16h an = a, bn0 = b0, bn1 = b1, bn2 = b2, bn3 = b3;
    const int kn = kk + 32;
    if (kn < HD) {
      an  = load_a_frag(arow + kn, hi);
      bn0 = load_b_frag(bcol0 + kn);
      bn1 = load_b_frag(bcol0 + 16 * HD + kn);
      bn2 = load_b_frag(bcol0 + 32 * HD + kn);
      bn3 = load_b_frag(bcol0 + 48 * HD + kn);
    }
    acc[0] = WMMA_F16(a, b0, acc[0]);
    acc[1] = WMMA_F16(a, b1, acc[1]);
    acc[2] = WMMA_F16(a, b2, acc[2]);
    acc[3] = WMMA_F16(a, b3, acc[3]);
    a = an; b0 = bn0; b1 = bn1; b2 = bn2; b3 = bn3;
  }
#pragma unroll
  for (int g = 0; g < 4; ++g)
#pragma unroll
    for (int r = 0; r < 8; ++r)
      out[(long)(row0 + r + hi * 8) * Cc + col0 + g * 16 + ln] = acc[g][r];
}

// ---------------------------------------------------------------------------
extern "C" void kernel_launch(void* const* d_in, const int* in_sizes, int n_in,
                              void* d_out, int out_size, void* d_ws, size_t ws_size,
                              hipStream_t stream) {
  const float* x        = (const float*)d_in[0];   // [B,N,C]
  const float* w_qkv    = (const float*)d_in[1];   // [C, 3HD]
  const float* w_out    = (const float*)d_in[2];   // [HD, C]
  const float* pos_bias = (const float*)d_in[3];   // [H,N,N]
  const float* temp     = (const float*)d_in[4];   // scalar
  const unsigned char* mask = (const unsigned char*)d_in[5]; // [B,N] bool
  float* out = (float*)d_out;

  char* p = (char*)d_ws;
  auto carve = [&](size_t bytes) { void* r = (void*)p; p += (bytes + 255) & ~(size_t)255; return r; };

  const long BN = (long)Bb * Nn;                   // 4096
  _Float16* xh    = (_Float16*)carve(BN * Cc * sizeof(_Float16));
  _Float16* wqkvT = (_Float16*)carve((long)QKVC * Cc * sizeof(_Float16));
  _Float16* woutT = (_Float16*)carve((long)Cc * HD * sizeof(_Float16));
  float*    Qf    = (float*)   carve(BN * HD * sizeof(float));
  float*    Kf    = (float*)   carve(BN * HD * sizeof(float));
  _Float16* Qh    = (_Float16*)carve(BN * HD * sizeof(_Float16));
  _Float16* Kh    = (_Float16*)carve(BN * HD * sizeof(_Float16));
  _Float16* Vt    = (_Float16*)carve(BN * HD * sizeof(_Float16));
  _Float16* Ah    = (_Float16*)carve(BN * HD * sizeof(_Float16));

  // 1. precision conversion / weight transposes
  cvt_f32_f16<<<2048, 256, 0, stream>>>(x, xh, BN * Cc);
  transpose_f32_f16<<<(int)((long)Cc * QKVC / 256), 256, 0, stream>>>(w_qkv, wqkvT, Cc, QKVC);
  transpose_f32_f16<<<(int)((long)HD * Cc / 256), 256, 0, stream>>>(w_out, woutT, HD, Cc);

  // 2. QKV projection with de-interleave (V stored transposed)
  qkv_gemm<<<1536, 128, 0, stream>>>(xh, wqkvT, Qf, Kf, Vt);

  // 3. L2 normalize, fold temperature into Q
  norm_qk<<<4096, 256, 0, stream>>>(Qf, Kf, temp, Qh, Kh);

  // 4. flash attention (TDM-streamed bias)
  attn_kernel<<<512, 128, 0, stream>>>(Qh, Kh, Vt, pos_bias, mask, Ah);

  // 5. output projection
  out_gemm<<<512, 128, 0, stream>>>(Ah, woutT, out);
}
